// RBFEncoder_45749991636996
// MI455X (gfx1250) — matrix operand order repaired and checked by reading, not compiled, not run on hardware
//
#include <hip/hip_runtime.h>

// ---------------------------------------------------------------------------
// RBF feature encoder for MI455X (gfx1250, wave32, WMMA).
//   out[n, g*512 + k] = exp(-gamma[g] * max(0, |x_n|^2 + |c_gk|^2 - 2 x_n.c_gk))
// GEMM core in bf16x3 split precision on v_wmma_f32_16x16x32_bf16.
// ---------------------------------------------------------------------------

typedef __attribute__((ext_vector_type(16))) __bf16 v16bf;
typedef __attribute__((ext_vector_type(8)))  float  v8f;

#define N_TOT   65536
#define D_DIM   64
#define G_CNT   4
#define K_CNT   512
#define OUTW    (G_CNT * K_CNT)     // 2048
#define TILE    128
#define LDS_STRIDE 36               // dwords per LDS row (72 bf16: 64 data + 8 pad)

union FragBF {
    uint4 u[2];
    v16bf v;
};

__device__ __forceinline__ unsigned int f2bf_rne(float f) {
    unsigned int u = __float_as_uint(f);
    return (u + 0x7FFFu + ((u >> 16) & 1u)) >> 16;   // round-to-nearest-even
}
__device__ __forceinline__ float bf2f(unsigned int h) {
    return __uint_as_float(h << 16);
}

__global__ __launch_bounds__(256) void rbf_encoder_wmma_bf16x3(
    const float* __restrict__ states,    // [N, 64]
    const float* __restrict__ centers,   // [G, 512, 64]
    const float* __restrict__ gammas,    // [G]
    float* __restrict__ out)             // [N, 2048]
{
    __shared__ unsigned int sAhi[TILE * LDS_STRIDE];
    __shared__ unsigned int sAlo[TILE * LDS_STRIDE];
    __shared__ unsigned int sBhi[TILE * LDS_STRIDE];
    __shared__ unsigned int sBlo[TILE * LDS_STRIDE];
    __shared__ float x2s[TILE];          // |x|^2 per tile row
    __shared__ float c2s[TILE];          // |c|^2 per tile column

    const int tid     = threadIdx.x;
    const int bx      = blockIdx.x;      // 16 column tiles (4 per gamma bank)
    const int by      = blockIdx.y;      // 512 row tiles
    const int g       = bx >> 2;
    const int kb0     = (bx & 3) * TILE; // center offset within bank g
    const int rowBase = by * TILE;

    if (tid < TILE) x2s[tid] = 0.0f;
    else            c2s[tid - TILE] = 0.0f;
    __syncthreads();

    const float* __restrict__ cbank = centers + (size_t)(g * K_CNT + kb0) * D_DIM;

    // ---- Stage: fp32 -> (hi,lo) bf16 into LDS once per WG; row norms via ds_add_f32
    #pragma unroll
    for (int which = 0; which < 2; ++which) {
        const float*  src  = which ? cbank : (states + (size_t)rowBase * D_DIM);
        unsigned int* dhi  = which ? sBhi : sAhi;
        unsigned int* dlo  = which ? sBlo : sAlo;
        float*        nrm  = which ? c2s  : x2s;
        for (int r = tid; r < TILE * 4; r += 256) {          // 16-element runs
            const int row = r >> 2;
            const int cs  = (r & 3) * 16;
            const float4* p = (const float4*)(src + (size_t)row * D_DIM + cs);
            float4 q0 = p[0], q1 = p[1], q2 = p[2], q3 = p[3];
            const float xs[16] = {q0.x,q0.y,q0.z,q0.w, q1.x,q1.y,q1.z,q1.w,
                                  q2.x,q2.y,q2.z,q2.w, q3.x,q3.y,q3.z,q3.w};
            unsigned int ph[8], pl[8];
            float ss = 0.0f;
            #pragma unroll
            for (int j = 0; j < 8; ++j) {
                float a0 = xs[2*j], a1 = xs[2*j+1];
                ss = fmaf(a0, a0, ss);
                ss = fmaf(a1, a1, ss);
                unsigned int h0 = f2bf_rne(a0), h1 = f2bf_rne(a1);
                unsigned int l0 = f2bf_rne(a0 - bf2f(h0));
                unsigned int l1 = f2bf_rne(a1 - bf2f(h1));
                ph[j] = h0 | (h1 << 16);
                pl[j] = l0 | (l1 << 16);
            }
            const unsigned int base = row * LDS_STRIDE + (cs >> 1);
            *(uint4*)&dhi[base]     = make_uint4(ph[0], ph[1], ph[2], ph[3]);
            *(uint4*)&dhi[base + 4] = make_uint4(ph[4], ph[5], ph[6], ph[7]);
            *(uint4*)&dlo[base]     = make_uint4(pl[0], pl[1], pl[2], pl[3]);
            *(uint4*)&dlo[base + 4] = make_uint4(pl[4], pl[5], pl[6], pl[7]);
            atomicAdd(&nrm[row], ss);                        // ds_add_f32
        }
    }
    __syncthreads();

    // ---- Compute: each wave owns a 16-row band, sweeps 8 B tiles of 16 cols
    const int wave = tid >> 5;           // 0..7 (wave32)
    const int lane = tid & 31;
    const int half = lane >> 4;          // 0: lanes 0-15, 1: lanes 16-31
    const int lm   = lane & 15;
    const int mloc = wave * 16;

    // A fragments (ISA 16-bit A 16x32 layout): row = lm for both halves;
    // half 0 holds K kb..kb+7 / kb+16..kb+23 with kb = half*8.
    FragBF aH[2], aL[2];
    {
        const int mrow = mloc + lm;
        const int kb   = half * 8;
        #pragma unroll
        for (int kc = 0; kc < 2; ++kc) {
            const unsigned int c0 = (unsigned)(kc * 32 + kb) >> 1;       // dword col
            const unsigned int c1 = (unsigned)(kc * 32 + 16 + kb) >> 1;
            aH[kc].u[0] = *(const uint4*)&sAhi[mrow * LDS_STRIDE + c0];
            aH[kc].u[1] = *(const uint4*)&sAhi[mrow * LDS_STRIDE + c1];
            aL[kc].u[0] = *(const uint4*)&sAlo[mrow * LDS_STRIDE + c0];
            aL[kc].u[1] = *(const uint4*)&sAlo[mrow * LDS_STRIDE + c1];
        }
    }

    // C layout: VGPR r -> M = r + half*8, N = lm
    float x2r[8];
    #pragma unroll
    for (int r = 0; r < 8; ++r) x2r[r] = x2s[mloc + half * 8 + r];

    const float gam = gammas[g];
    float* outp = out + (size_t)(rowBase + mloc + half * 8) * OUTW
                      + (size_t)g * K_CNT + kb0 + lm;

    for (int t = 0; t < 8; ++t) {
        // B fragments (ISA 16-bit B 32x16 layout): lane = column t*16+lm,
        // halves split K: lanes 0-15 K 0..7/16..23, lanes 16-31 K 8..15/24..31 per chunk.
        const int nrow = t * 16 + lm;
        FragBF bH[2], bL[2];
        #pragma unroll
        for (int kc = 0; kc < 2; ++kc) {
            const unsigned int c0 = (unsigned)(kc * 32 + half * 8) >> 1;
            bH[kc].u[0] = *(const uint4*)&sBhi[nrow * LDS_STRIDE + c0];
            bH[kc].u[1] = *(const uint4*)&sBhi[nrow * LDS_STRIDE + c0 + 8];
            bL[kc].u[0] = *(const uint4*)&sBlo[nrow * LDS_STRIDE + c0];
            bL[kc].u[1] = *(const uint4*)&sBlo[nrow * LDS_STRIDE + c0 + 8];
        }
        const float c2n = c2s[nrow];

        // bf16x3 split: hi*hi + hi*lo + lo*hi (lo*lo dropped, ~2^-34 abs)
        v8f acc = {};
        #pragma unroll
        for (int kc = 0; kc < 2; ++kc) {
            acc = __builtin_amdgcn_wmma_f32_16x16x32_bf16(
                      false, aH[kc].v, false, bH[kc].v, (short)0, acc, false, false);
            acc = __builtin_amdgcn_wmma_f32_16x16x32_bf16(
                      false, aH[kc].v, false, bL[kc].v, (short)0, acc, false, false);
            acc = __builtin_amdgcn_wmma_f32_16x16x32_bf16(
                      false, aL[kc].v, false, bH[kc].v, (short)0, acc, false, false);
        }

        #pragma unroll
        for (int r = 0; r < 8; ++r) {
            float d2 = fmaf(-2.0f, acc[r], x2r[r] + c2n);
            d2 = fmaxf(d2, 0.0f);
            const float f = __expf(-gam * d2);
            // write-once 537 MB stream: keep it out of L2 with NT stores
            __builtin_nontemporal_store(f, outp + (size_t)r * OUTW + t * 16);
        }
    }
}

extern "C" void kernel_launch(void* const* d_in, const int* in_sizes, int n_in,
                              void* d_out, int out_size, void* d_ws, size_t ws_size,
                              hipStream_t stream) {
    (void)in_sizes; (void)n_in; (void)out_size; (void)d_ws; (void)ws_size;
    const float* states  = (const float*)d_in[0];
    const float* centers = (const float*)d_in[1];
    const float* gammas  = (const float*)d_in[2];
    float* out = (float*)d_out;

    dim3 grid(OUTW / TILE, N_TOT / TILE, 1);   // 16 x 512 workgroups
    dim3 block(256, 1, 1);                     // 8 wave32 waves
    rbf_encoder_wmma_bf16x3<<<grid, block, 0, stream>>>(states, centers, gammas, out);
}
